// FuseLayer_21199958573445
// MI455X (gfx1250) — matrix operand order repaired and checked by reading, not compiled, not run on hardware
//
#include <hip/hip_runtime.h>
#include <cstddef>

// ---------------- types ----------------
typedef __bf16 v16bf __attribute__((ext_vector_type(16)));
typedef __bf16 v8bf  __attribute__((ext_vector_type(8)));
typedef float  v8f   __attribute__((ext_vector_type(8)));

static __device__ __forceinline__ v16bf cat8(v8bf lo, v8bf hi) {
  return __builtin_shufflevector(lo, hi, 0,1,2,3,4,5,6,7,8,9,10,11,12,13,14,15);
}

#define DIMF 128
#define NCOL 384   // 3 * DIM concatenated feature blocks

// ---------------- prep kernels ----------------

__global__ void init_inv_kernel(int* inv0, int* inv1) {
  int t = blockIdx.x * 256 + threadIdx.x;
  if (t < 8000) inv0[t] = -1;
  if (t < 4000) inv1[t] = -1;
}

__global__ void scatter_inv_kernel(const int* __restrict__ idx0, const int* __restrict__ idx1,
                                   int* inv0, int* inv1) {
  int t = blockIdx.x * 256 + threadIdx.x;
  if (t < 4000) inv0[idx0[t]] = t;
  if (t < 2000) inv1[idx1[t]] = t;
}

// Wt[(i,j)][n*128+k] = Ws[(i,j)][k*128+n]  (bf16, N-major for B fragments)
__global__ void build_wt_kernel(const float* __restrict__ Ws, __bf16* __restrict__ Wt) {
  int t = blockIdx.x * 256 + threadIdx.x;
  if (t >= 9 * 16384) return;
  int ij = t >> 14, rem = t & 16383;
  int n = rem >> 7, k = rem & 127;
  Wt[(size_t)ij * 16384 + n * 128 + k] = (__bf16)Ws[(size_t)ij * 16384 + k * 128 + n];
}

// Build Xt (NCOL x Ni, node-contiguous) in bf16 with gather/scatter chains.
__global__ void build_xt_kernel(int level, int Ni,
                                const float* __restrict__ h0, const float* __restrict__ h1,
                                const float* __restrict__ h2,
                                const int* __restrict__ idx0, const int* __restrict__ idx1,
                                const int* __restrict__ inv0, const int* __restrict__ inv1,
                                __bf16* __restrict__ Xt) {
  long t = (long)blockIdx.x * 256 + threadIdx.x;
  if (t >= (long)Ni * NCOL) return;
  int r = (int)(t % Ni);
  int n = (int)(t / Ni);
  int j = n >> 7;          // source level
  int f = n & 127;         // feature
  float v = 0.f;
  if (level == 0) {
    if (j == 0) v = h0[(size_t)r * DIMF + f];
    else if (j == 1) { int q = inv0[r]; if (q >= 0) v = h1[(size_t)q * DIMF + f]; }
    else { int q = inv0[r];
           if (q >= 0) { int p = inv1[q]; if (p >= 0) v = h2[(size_t)p * DIMF + f]; } }
  } else if (level == 1) {
    if (j == 0) v = h0[(size_t)idx0[r] * DIMF + f];
    else if (j == 1) v = h1[(size_t)r * DIMF + f];
    else { int p = inv1[r]; if (p >= 0) v = h2[(size_t)p * DIMF + f]; }
  } else {
    if (j == 0) v = h0[(size_t)idx0[idx1[r]] * DIMF + f];
    else if (j == 1) v = h1[(size_t)idx1[r] * DIMF + f];
    else v = h2[(size_t)r * DIMF + f];
  }
  Xt[(size_t)n * Ni + r] = (__bf16)v;
}

// ---------------- pass A: Y = adj @ X  (bf16 WMMA, f32 acc) ----------------
// 512 threads = 16 waves.  BM=128 rows, BN=384 cols (full), K-loop step 32.
// Register-staged pipeline: tile k+1 loaded from global while tile k computes.
// Row bounds handled by clamping once (garbage rows never stored); K remainder
// (only level 2: 2000 = 62*32+16) peeled into a guarded epilogue step.

#define ALD_STRIDE 40   // bf16 elems per row (32 data + 8 pad) -> 80B, 16B aligned
__global__ __launch_bounds__(512, 1)
void gemm_adj_kernel(int Ni, const float* __restrict__ adj,
                     const __bf16* __restrict__ Xt, __bf16* __restrict__ Y) {
  __shared__ __bf16 Alds[128 * ALD_STRIDE];
  __shared__ __bf16 Blds[NCOL * ALD_STRIDE];

  const int tid   = threadIdx.x;
  const int lane  = tid & 31;
  const int wave  = tid >> 5;
  const int mwave = wave & 7;
  const int nhalf = wave >> 3;
  const int mBase = blockIdx.x * 128 + mwave * 16;

  // ---- loader thread mapping (loop-invariant) ----
  // A: 4 threads per row, 8 floats each
  const int arow_ld = tid >> 2;
  const int akq     = (tid & 3) * 8;
  int grow = blockIdx.x * 128 + arow_ld;
  if (grow >= Ni) grow = Ni - 1;                 // clamp once; junk rows never stored
  const float* aptr = adj + (size_t)grow * Ni + akq;
  // B: 1536 16B chunks / 512 threads (3 each)
  int bn0 = (tid) >> 2,            bk0 = ((tid) & 3) * 8;
  int bn1 = (tid + 512) >> 2,      bk1 = ((tid + 512) & 3) * 8;
  int bn2 = (tid + 1024) >> 2,     bk2 = ((tid + 1024) & 3) * 8;
  const __bf16* bptr0 = Xt + (size_t)bn0 * Ni + bk0;
  const __bf16* bptr1 = Xt + (size_t)bn1 * Ni + bk1;
  const __bf16* bptr2 = Xt + (size_t)bn2 * Ni + bk2;

  // ---- compute thread mapping (loop-invariant LDS offsets) ----
  const int arow = mwave * 16 + (lane & 15);
  const int aoff = (lane >> 4) * 8;       // K granule 8 for A fragments
  const int boff = (lane >> 4) * 8;       // K granule 8 for B fragments
  const __bf16* aldsP = &Alds[arow * ALD_STRIDE + aoff];
  const __bf16* bldsP = &Blds[(nhalf * 192 + (lane & 15)) * ALD_STRIDE + boff];

  v8f acc[12] = {};

  const int KSfull = Ni >> 5;       // full 32-wide K steps
  const int krem   = Ni & 31;       // leftover K (0 or 16 here)

  v8bf areg, breg0, breg1, breg2;
  // prologue: load tile 0
  {
    const float4* p = (const float4*)aptr;
    float4 x0 = p[0], x1 = p[1];
    areg[0] = (__bf16)x0.x; areg[1] = (__bf16)x0.y; areg[2] = (__bf16)x0.z; areg[3] = (__bf16)x0.w;
    areg[4] = (__bf16)x1.x; areg[5] = (__bf16)x1.y; areg[6] = (__bf16)x1.z; areg[7] = (__bf16)x1.w;
    breg0 = *(const v8bf*)bptr0;
    breg1 = *(const v8bf*)bptr1;
    breg2 = *(const v8bf*)bptr2;
    aptr += 32; bptr0 += 32; bptr1 += 32; bptr2 += 32;
  }

  for (int ks = 0; ks < KSfull; ++ks) {
    // stage current tile into LDS
    *(v8bf*)&Alds[arow_ld * ALD_STRIDE + akq] = areg;
    *(v8bf*)&Blds[bn0 * ALD_STRIDE + bk0] = breg0;
    *(v8bf*)&Blds[bn1 * ALD_STRIDE + bk1] = breg1;
    *(v8bf*)&Blds[bn2 * ALD_STRIDE + bk2] = breg2;
    __syncthreads();
    // issue next tile's global loads (latency overlapped with WMMAs below)
    if (ks + 1 < KSfull) {
      const float4* p = (const float4*)aptr;
      float4 x0 = p[0], x1 = p[1];
      areg[0] = (__bf16)x0.x; areg[1] = (__bf16)x0.y; areg[2] = (__bf16)x0.z; areg[3] = (__bf16)x0.w;
      areg[4] = (__bf16)x1.x; areg[5] = (__bf16)x1.y; areg[6] = (__bf16)x1.z; areg[7] = (__bf16)x1.w;
      breg0 = *(const v8bf*)bptr0;
      breg1 = *(const v8bf*)bptr1;
      breg2 = *(const v8bf*)bptr2;
      aptr += 32; bptr0 += 32; bptr1 += 32; bptr2 += 32;
    }
    // compute: 1 A fragment, 12 B fragments, 12 WMMAs
    {
      v8bf alo = *(const v8bf*)(aldsP);
      v8bf ahi = *(const v8bf*)(aldsP + 16);
      v16bf af = cat8(alo, ahi);
      #pragma unroll
      for (int nt = 0; nt < 12; ++nt) {
        const __bf16* bp = bldsP + nt * 16 * ALD_STRIDE;
        v8bf blo = *(const v8bf*)(bp);
        v8bf bhi = *(const v8bf*)(bp + 16);
        v16bf bf = cat8(blo, bhi);
        acc[nt] = __builtin_amdgcn_wmma_f32_16x16x32_bf16(
            false, af, false, bf, (short)0, acc[nt], false, false);
      }
    }
    __syncthreads();
  }

  // peeled K remainder (guarded, runs at most once)
  if (krem) {
    const int k0 = KSfull * 32;
    v8bf av;
    #pragma unroll
    for (int e = 0; e < 8; ++e) {
      int kk = k0 + akq + e; if (kk >= Ni) kk = Ni - 1;   // junk * 0 (B zeroed)
      av[e] = (__bf16)adj[(size_t)grow * Ni + kk];
    }
    *(v8bf*)&Alds[arow_ld * ALD_STRIDE + akq] = av;
    v8bf bv0, bv1, bv2;
    #pragma unroll
    for (int e = 0; e < 8; ++e) {
      int kg0 = k0 + bk0 + e, kg1 = k0 + bk1 + e, kg2 = k0 + bk2 + e;
      bv0[e] = (kg0 < Ni) ? Xt[(size_t)bn0 * Ni + kg0] : (__bf16)0.f;
      bv1[e] = (kg1 < Ni) ? Xt[(size_t)bn1 * Ni + kg1] : (__bf16)0.f;
      bv2[e] = (kg2 < Ni) ? Xt[(size_t)bn2 * Ni + kg2] : (__bf16)0.f;
    }
    *(v8bf*)&Blds[bn0 * ALD_STRIDE + bk0] = bv0;
    *(v8bf*)&Blds[bn1 * ALD_STRIDE + bk1] = bv1;
    *(v8bf*)&Blds[bn2 * ALD_STRIDE + bk2] = bv2;
    __syncthreads();
    v8bf alo = *(const v8bf*)(aldsP);
    v8bf ahi = *(const v8bf*)(aldsP + 16);
    v16bf af = cat8(alo, ahi);
    #pragma unroll
    for (int nt = 0; nt < 12; ++nt) {
      const __bf16* bp = bldsP + nt * 16 * ALD_STRIDE;
      v8bf blo = *(const v8bf*)(bp);
      v8bf bhi = *(const v8bf*)(bp + 16);
      v16bf bf = cat8(blo, bhi);
      acc[nt] = __builtin_amdgcn_wmma_f32_16x16x32_bf16(
          false, af, false, bf, (short)0, acc[nt], false, false);
    }
  }

  // ---- store Y (bf16, row-major stride NCOL)
  #pragma unroll
  for (int nt = 0; nt < 12; ++nt) {
    int col = nhalf * 192 + nt * 16 + (lane & 15);
    #pragma unroll
    for (int v = 0; v < 8; ++v) {
      int row = mBase + v + ((lane >> 4) * 8);
      if (row < Ni) Y[(size_t)row * NCOL + col] = (__bf16)acc[nt][v];
    }
  }
}

// ---------------- pass B: out = sum_j relu(Y[:,j] @ W[i,j] + b[i,j]) ----------------
// 256 threads = 8 waves; BM=128 (16 rows/wave), BN=128 (8 tiles), K=128 per j.
__global__ __launch_bounds__(256, 1)
void gemm_proj_kernel(int Ni, const __bf16* __restrict__ Y,
                      const __bf16* __restrict__ Wt,    // 3 * 128*128, level i's slice
                      const float* __restrict__ bvec,   // 3 * 128
                      float* __restrict__ out) {
  const int tid  = threadIdx.x;
  const int lane = tid & 31;
  const int wave = tid >> 5;
  const int mBase = blockIdx.x * 128 + wave * 16;

  v8f outac[8] = {};
  int yrow = mBase + (lane & 15);
  if (yrow >= Ni) yrow = Ni - 1;      // clamped rows never stored

  #pragma unroll
  for (int j = 0; j < 3; ++j) {
    v8f cur[8] = {};
    const __bf16* W = Wt + (size_t)j * 16384;
    #pragma unroll
    for (int ks = 0; ks < 4; ++ks) {
      int k0   = j * DIMF + ks * 32;
      int aoff = (lane >> 4) * 8;
      v8bf alo = *(const v8bf*)(Y + (size_t)yrow * NCOL + k0 + aoff);
      v8bf ahi = *(const v8bf*)(Y + (size_t)yrow * NCOL + k0 + aoff + 16);
      v16bf af = cat8(alo, ahi);
      int kk0  = ks * 32;
      #pragma unroll
      for (int nt = 0; nt < 8; ++nt) {
        int n    = nt * 16 + (lane & 15);
        int boff = (lane >> 4) * 8;
        v8bf blo = *(const v8bf*)(W + (size_t)n * DIMF + kk0 + boff);
        v8bf bhi = *(const v8bf*)(W + (size_t)n * DIMF + kk0 + boff + 16);
        v16bf bf = cat8(blo, bhi);
        cur[nt] = __builtin_amdgcn_wmma_f32_16x16x32_bf16(
            false, af, false, bf, (short)0, cur[nt], false, false);
      }
    }
    #pragma unroll
    for (int nt = 0; nt < 8; ++nt) {
      float bval = bvec[j * DIMF + nt * 16 + (lane & 15)];
      #pragma unroll
      for (int v = 0; v < 8; ++v) {
        float x = cur[nt][v] + bval;
        outac[nt][v] += (x > 0.f) ? x : 0.f;
      }
    }
  }
  #pragma unroll
  for (int nt = 0; nt < 8; ++nt) {
    int col = nt * 16 + (lane & 15);
    #pragma unroll
    for (int v = 0; v < 8; ++v) {
      int row = mBase + v + ((lane >> 4) * 8);
      if (row < Ni) out[(size_t)row * DIMF + col] = outac[nt][v];
    }
  }
}

// ---------------- host launcher ----------------
extern "C" void kernel_launch(void* const* d_in, const int* in_sizes, int n_in,
                              void* d_out, int out_size, void* d_ws, size_t ws_size,
                              hipStream_t stream) {
  (void)in_sizes; (void)n_in; (void)out_size; (void)ws_size;
  const int N0 = 8000, N1 = 4000, N2 = 2000;
  const float* adj0 = (const float*)d_in[0];
  const float* adj1 = (const float*)d_in[1];
  const float* adj2 = (const float*)d_in[2];
  const float* h0   = (const float*)d_in[3];
  const float* h1   = (const float*)d_in[4];
  const float* h2   = (const float*)d_in[5];
  const int*   idx0 = (const int*)d_in[6];
  const int*   idx1 = (const int*)d_in[7];
  const float* Ws   = (const float*)d_in[8];
  const float* bs   = (const float*)d_in[9];

  // workspace carve (256B aligned)
  char* base = (char*)d_ws;
  size_t off = 0;
  auto carve = [&](size_t bytes) -> char* {
    off = (off + 255) & ~(size_t)255;
    char* p = base + off;
    off += bytes;
    return p;
  };
  __bf16* Xt0 = (__bf16*)carve((size_t)NCOL * N0 * 2);
  __bf16* Xt1 = (__bf16*)carve((size_t)NCOL * N1 * 2);
  __bf16* Xt2 = (__bf16*)carve((size_t)NCOL * N2 * 2);
  __bf16* Y0  = (__bf16*)carve((size_t)N0 * NCOL * 2);
  __bf16* Y1  = (__bf16*)carve((size_t)N1 * NCOL * 2);
  __bf16* Y2  = (__bf16*)carve((size_t)N2 * NCOL * 2);
  __bf16* Wt  = (__bf16*)carve((size_t)9 * 16384 * 2);
  int*    inv0 = (int*)carve((size_t)N0 * 4);
  int*    inv1 = (int*)carve((size_t)N1 * 4);

  float* out0 = (float*)d_out;
  float* out1 = out0 + (size_t)N0 * DIMF;
  float* out2 = out1 + (size_t)N1 * DIMF;

  // prep
  init_inv_kernel<<<(N0 + 255) / 256, 256, 0, stream>>>(inv0, inv1);
  scatter_inv_kernel<<<(N1 + 255) / 256, 256, 0, stream>>>(idx0, idx1, inv0, inv1);
  build_wt_kernel<<<(9 * 16384 + 255) / 256, 256, 0, stream>>>(Ws, Wt);
  build_xt_kernel<<<(int)(((long)N0 * NCOL + 255) / 256), 256, 0, stream>>>(
      0, N0, h0, h1, h2, idx0, idx1, inv0, inv1, Xt0);
  build_xt_kernel<<<(int)(((long)N1 * NCOL + 255) / 256), 256, 0, stream>>>(
      1, N1, h0, h1, h2, idx0, idx1, inv0, inv1, Xt1);
  build_xt_kernel<<<(int)(((long)N2 * NCOL + 255) / 256), 256, 0, stream>>>(
      2, N2, h0, h1, h2, idx0, idx1, inv0, inv1, Xt2);

  // pass A: Y_i = adj_i @ X_i
  gemm_adj_kernel<<<(N0 + 127) / 128, 512, 0, stream>>>(N0, adj0, Xt0, Y0);
  gemm_adj_kernel<<<(N1 + 127) / 128, 512, 0, stream>>>(N1, adj1, Xt1, Y1);
  gemm_adj_kernel<<<(N2 + 127) / 128, 512, 0, stream>>>(N2, adj2, Xt2, Y2);

  // pass B: out_i = sum_j relu(Y_i[:,j] @ W[i,j] + b[i,j])
  gemm_proj_kernel<<<(N0 + 127) / 128, 256, 0, stream>>>(N0, Y0, Wt + 0 * 3 * 16384, bs + 0 * 3 * DIMF, out0);
  gemm_proj_kernel<<<(N1 + 127) / 128, 256, 0, stream>>>(N1, Y1, Wt + 1 * 3 * 16384, bs + 1 * 3 * DIMF, out1);
  gemm_proj_kernel<<<(N2 + 127) / 128, 256, 0, stream>>>(N2, Y2, Wt + 2 * 3 * 16384, bs + 2 * 3 * DIMF, out2);
}
